// TABL_25340307046630
// MI455X (gfx1250) — matrix operand order repaired and checked by reading, not compiled, not run on hardware
//
#include <hip/hip_runtime.h>

// TABL fused forward for MI455X (gfx1250, wave32, WMMA + async LDS DMA).
// Workspace (~66 MB):
//   [0,    512K) : Wm^T bf16 [c][k], diag fixed to 1/512      512*512*2
//   [512K, 768K) : W2^T bf16 [c][k]                           256*512*2
//   [768K,1792K) : W1^T bf16 [m][k]                           512*1024*2
//   [1792K,+64M) : h bf16 [N][d1][D2]                         128*512*512*2

typedef __attribute__((ext_vector_type(16))) __bf16 v16bf;
typedef __attribute__((ext_vector_type(8)))  __bf16 v8bf;
typedef __attribute__((ext_vector_type(8)))  float  v8f;

constexpr int N_  = 128;
constexpr int D1_ = 1024;
constexpr int D2_ = 512;
constexpr int d1_ = 512;
constexpr int d2_ = 256;

// ---------------- gfx1250 async global->LDS DMA (ASYNCcnt) ----------------
typedef __attribute__((address_space(3))) void lds_t;

__device__ inline void async_load_b128(void* lds_dst, const void* gsrc) {
  unsigned l = (unsigned)(size_t)(lds_t*)lds_dst;           // raw LDS byte offset
  unsigned long long g = (unsigned long long)(size_t)gsrc;  // 64-bit global addr
  asm volatile("global_load_async_to_lds_b128 %0, %1, off" :: "v"(l), "v"(g) : "memory");
}
__device__ inline void async_wait0() { asm volatile("s_wait_asynccnt 0" ::: "memory"); }
__device__ inline void async_wait2() { asm volatile("s_wait_asynccnt 2" ::: "memory"); }

// ---------------- WMMA fragment helpers (ISA §7.12.2, wave32) ----------------
// Tiles stored K-contiguous: A as [m][ldk], B as [c][ldk]. Every fragment is
// two 16B LDS loads (ds_load_b128) instead of 16 scalar u16 loads.
union FragBF { v16bf v; struct { v8bf lo, hi; } h; };

// A lane (row m, half hf): e<8 -> k = k0+8*hf+e ; e>=8 -> k = k0+16+8*hf+(e-8)
__device__ inline v16bf frag_a(const __bf16* rows, int ldk, int m, int k0, int hf) {
  FragBF f;
  f.h.lo = *(const v8bf*)&rows[m * ldk + k0 + 8 * hf];
  f.h.hi = *(const v8bf*)&rows[m * ldk + k0 + 16 + 8 * hf];
  return f.v;
}
// B lane (col c, half hf): e<8 -> k = k0+8*hf+e ; e>=8 -> k = k0+16+8*hf+(e-8)
__device__ inline v16bf frag_b(const __bf16* cols, int ldk, int c, int k0, int hf) {
  FragBF f;
  f.h.lo = *(const v8bf*)&cols[c * ldk + k0 + 8 * hf];
  f.h.hi = *(const v8bf*)&cols[c * ldk + k0 + 16 + 8 * hf];
  return f.v;
}
__device__ inline v8f wmma_bf16(v16bf a, v16bf b, v8f c) {
  return __builtin_amdgcn_wmma_f32_16x16x32_bf16(false, a, false, b, (short)0, c, false, false);
}
__device__ inline v8f vzero() {
  v8f z;
#pragma unroll
  for (int i = 0; i < 8; ++i) z[i] = 0.0f;
  return z;
}
__device__ inline unsigned pack2bf(float x0, float x1) {
  union { __bf16 b[2]; unsigned u; } t;
  t.b[0] = (__bf16)x0; t.b[1] = (__bf16)x1;   // -> v_cvt_pk_bf16_f32
  return t.u;
}

// Stage a 128col x 32k tile of a pre-transposed bf16 weight [c][ldk] via async DMA.
__device__ inline void stage_tileT(const __bf16* __restrict__ srcT, int ldk,
                                   int cbase, int k0, __bf16* buf, int tid) {
#pragma unroll
  for (int i = 0; i < 2; ++i) {           // 512 x b128 per tile, 2 per thread
    int u = tid + i * 256;
    int c = u >> 2, q = (u & 3) * 8;
    async_load_b128(&buf[c * 32 + q], &srcT[(size_t)(cbase + c) * ldk + k0 + q]);
  }
}

// ---------------- Kernel 0: weight prep (f32 -> bf16, transpose, fix diag) ----------
__global__ __launch_bounds__(256) void tabl_prep_kernel(const float* __restrict__ W,
                                                        const float* __restrict__ W2,
                                                        const float* __restrict__ W1,
                                                        __bf16* __restrict__ WmT,
                                                        __bf16* __restrict__ W2T,
                                                        __bf16* __restrict__ W1T) {
  const int stride = gridDim.x * blockDim.x;
  const int t0 = blockIdx.x * blockDim.x + threadIdx.x;
  for (int i = t0; i < D2_ * D2_; i += stride) {          // WmT[c][k]
    int c = i >> 9, k = i & (D2_ - 1);
    float v = (k == c) ? (1.0f / (float)D2_) : W[(size_t)k * D2_ + c];
    WmT[i] = (__bf16)v;
  }
  for (int i = t0; i < d2_ * D2_; i += stride) {          // W2T[c][k]
    int c = i >> 9, k = i & (D2_ - 1);
    W2T[i] = (__bf16)W2[(size_t)k * d2_ + c];
  }
  for (int i = t0; i < d1_ * D1_; i += stride) {          // W1T[m][k]
    int m = i >> 10, k = i & (D1_ - 1);
    W1T[i] = (__bf16)W1[(size_t)k * d1_ + m];
  }
}

// ---------------- Kernel 1: h[n] = W1^T @ x[n] ----------------
// 128x128 tile, K-step 32. 8 waves: 4 along M (32 rows) x 2 along N (64 cols).
// A tile (W1T) arrives via async DMA; x tile is f32-loaded, cvt'd, transposed into [c][k].
__global__ __launch_bounds__(256) void tabl_gemm1_kernel(const float* __restrict__ x,
                                                         const __bf16* __restrict__ W1T,
                                                         __bf16* __restrict__ h) {
  __shared__ __align__(16) __bf16 As[128 * 32];  // [m][k]
  __shared__ __align__(16) __bf16 Bs[128 * 32];  // [c][k]
  const int tid  = threadIdx.x;
  const int lane = tid & 31, wave = tid >> 5;
  const int hf = lane >> 4, ln = lane & 15;
  const int wm = (wave & 3) * 32;
  const int wn = (wave >> 2) * 64;
  const int w0 = blockIdx.x * 128;  // D2 tile base
  const int m0 = blockIdx.y * 128;  // d1 tile base
  const int n  = blockIdx.z;

  v8f acc[2][4];
#pragma unroll
  for (int i = 0; i < 2; ++i)
#pragma unroll
    for (int j = 0; j < 4; ++j) acc[i][j] = vzero();

  for (int k0 = 0; k0 < D1_; k0 += 32) {
    // A tile: straight async copy of W1T rows m0..m0+127, k0..k0+31
    stage_tileT(W1T, D1_, m0, k0, As, tid);
    // B tile: x f32 rows (k0+2kk2, k0+2kk2+1) -> bf16 pairs into Bs[c][k]
#pragma unroll
    for (int i = 0; i < 2; ++i) {
      int u = tid + i * 256;               // 512 dual-row float4 units
      int kk2 = u >> 5, c4 = (u & 31) * 4;
      const float* r0 = &x[((size_t)n * D1_ + k0 + 2 * kk2) * D2_ + w0 + c4];
      const float* r1 = r0 + D2_;
      float4 v0 = *(const float4*)r0;
      float4 v1 = *(const float4*)r1;
      const float* p0 = (const float*)&v0;
      const float* p1 = (const float*)&v1;
#pragma unroll
      for (int j = 0; j < 4; ++j)
        *(unsigned*)&Bs[(c4 + j) * 32 + 2 * kk2] = pack2bf(p0[j], p1[j]);
    }
    if (k0 + 32 < D1_)
      __builtin_prefetch(&x[((size_t)n * D1_ + k0 + 32) * D2_ + w0 + (tid & 31) * 4], 0, 0);
    async_wait0();
    __syncthreads();

    v16bf af[2], bfv[4];
#pragma unroll
    for (int sm = 0; sm < 2; ++sm) af[sm]  = frag_a(As, 32, wm + sm * 16 + ln, 0, hf);
#pragma unroll
    for (int sn = 0; sn < 4; ++sn) bfv[sn] = frag_b(Bs, 32, wn + sn * 16 + ln, 0, hf);
#pragma unroll
    for (int sm = 0; sm < 2; ++sm)
#pragma unroll
      for (int sn = 0; sn < 4; ++sn) acc[sm][sn] = wmma_bf16(af[sm], bfv[sn], acc[sm][sn]);
    __syncthreads();
  }

#pragma unroll
  for (int sm = 0; sm < 2; ++sm)
#pragma unroll
    for (int sn = 0; sn < 4; ++sn)
#pragma unroll
      for (int r = 0; r < 8; ++r) {
        int row = m0 + wm + sm * 16 + r + 8 * hf;
        int col = w0 + wn + sn * 16 + ln;
        h[((size_t)n * d1_ + row) * D2_ + col] = (__bf16)acc[sm][sn][r];
      }
}

// ---------------- Kernel 2: fused  S = H@Wm, softmax, gate, Out = G@W2 + bias --------
// Per block: one n, 64 rows of d1. Dynamic LDS (~210 KB of the 320 KB WGP pool):
//   Hs: 64x512 bf16 (64K)  Ss: 64x512 f32 (128K)  Bs: 2x(128x32) bf16 (16K)  red (1K)
constexpr int FK_HS   = 64 * 512 * 2;
constexpr int FK_SS   = 64 * 512 * 4;
constexpr int FK_BS   = 2 * 128 * 32 * 2;
constexpr int FK_RED  = 64 * 4 * 4;
constexpr int FK_SMEM = FK_HS + FK_SS + FK_BS + FK_RED;

__global__ __launch_bounds__(256) void tabl_fused_kernel(const __bf16* __restrict__ h,
                                                         const __bf16* __restrict__ WmT,
                                                         const __bf16* __restrict__ W2T,
                                                         const float* __restrict__ alpha,
                                                         const float* __restrict__ bias,
                                                         float* __restrict__ out) {
  extern __shared__ __align__(16) char smem[];
  __bf16* Hs  = (__bf16*)smem;
  float*  Ss  = (float*)(smem + FK_HS);
  __bf16* Bs0 = (__bf16*)(smem + FK_HS + FK_SS);
  __bf16* Bs1 = Bs0 + 128 * 32;
  float*  red = (float*)(smem + FK_HS + FK_SS + FK_BS);

  const int tid  = threadIdx.x;
  const int lane = tid & 31, wave = tid >> 5;
  const int hf = lane >> 4, ln = lane & 15;
  const int m0 = blockIdx.x * 64;
  const int n  = blockIdx.y;
  const int wm = (wave & 3) * 16;   // 4 waves along M (16 rows each)
  const int wn = (wave >> 2) * 64;  // 2 waves along N (64 cols each)

  const __bf16* hrows = h + ((size_t)n * d1_ + m0) * D2_;

  // --- phase 0: async-DMA 64x512 bf16 H rows into LDS ---
#pragma unroll
  for (int i = 0; i < 16; ++i) {
    int idx4 = tid + i * 256;                 // 4096 x b128
    int row = idx4 >> 6, c8 = (idx4 & 63) * 8;
    async_load_b128(&Hs[row * 512 + c8], &hrows[row * 512 + c8]);
  }
  async_wait0();
  __syncthreads();

  // --- phase 1: S = H @ Wm (f32 scores kept entirely in LDS), double-buffered B ---
  for (int nc = 0; nc < 4; ++nc) {
    v8f acc[4];
#pragma unroll
    for (int j = 0; j < 4; ++j) acc[j] = vzero();
    stage_tileT(WmT, D2_, nc * 128, 0, Bs0, tid);
    for (int k0 = 0; k0 < D2_; k0 += 32) {
      int t = k0 >> 5;
      __bf16* cur = (t & 1) ? Bs1 : Bs0;
      __bf16* nxt = (t & 1) ? Bs0 : Bs1;
      if (k0 + 32 < D2_) { stage_tileT(WmT, D2_, nc * 128, k0 + 32, nxt, tid); async_wait2(); }
      else               { async_wait0(); }
      __syncthreads();
      v16bf a = frag_a(Hs, 512, wm + ln, k0, hf);
#pragma unroll
      for (int sn = 0; sn < 4; ++sn)
        acc[sn] = wmma_bf16(a, frag_b(cur, 32, wn + sn * 16 + ln, 0, hf), acc[sn]);
      __syncthreads();
    }
#pragma unroll
    for (int sn = 0; sn < 4; ++sn)
#pragma unroll
      for (int r = 0; r < 8; ++r)
        Ss[(wm + r + 8 * hf) * 512 + nc * 128 + wn + sn * 16 + ln] = acc[sn][r];
  }
  __syncthreads();

  // --- phase 2: row softmax + alpha-gating, G overwrites Hs (bf16) ---
  {
    const float ac = fminf(fmaxf(alpha[0], 0.0f), 10.0f);
    const int row = tid >> 2, part = tid & 3;   // 64 rows x 4 partitions of 128
    float* srow = Ss + row * 512 + part * 128;
    float mx = -3.4e38f;
    for (int i = 0; i < 128; ++i) mx = fmaxf(mx, srow[i]);
    red[row * 4 + part] = mx;
    __syncthreads();
    float rmx = fmaxf(fmaxf(red[row * 4 + 0], red[row * 4 + 1]),
                      fmaxf(red[row * 4 + 2], red[row * 4 + 3]));
    __syncthreads();
    float s = 0.0f;
    for (int i = 0; i < 128; ++i) {
      float e = __expf(srow[i] - rmx);
      srow[i] = e;
      s += e;
    }
    red[row * 4 + part] = s;
    __syncthreads();
    float inv = 1.0f / (red[row * 4 + 0] + red[row * 4 + 1] +
                        red[row * 4 + 2] + red[row * 4 + 3]);
    __bf16* grow = Hs + row * 512 + part * 128;
    for (int i = 0; i < 128; ++i) {
      float att = srow[i] * inv;
      float hv  = (float)grow[i];
      grow[i] = (__bf16)(ac * hv + (1.0f - ac) * hv * att);
    }
  }
  __syncthreads();

  // --- phase 3: Out = G @ W2 + bias (f32 store), double-buffered B ---
  for (int nc = 0; nc < 2; ++nc) {
    v8f acc[4];
#pragma unroll
    for (int j = 0; j < 4; ++j) acc[j] = vzero();
    stage_tileT(W2T, D2_, nc * 128, 0, Bs0, tid);
    for (int k0 = 0; k0 < D2_; k0 += 32) {
      int t = k0 >> 5;
      __bf16* cur = (t & 1) ? Bs1 : Bs0;
      __bf16* nxt = (t & 1) ? Bs0 : Bs1;
      if (k0 + 32 < D2_) { stage_tileT(W2T, D2_, nc * 128, k0 + 32, nxt, tid); async_wait2(); }
      else               { async_wait0(); }
      __syncthreads();
      v16bf a = frag_a(Hs, 512, wm + ln, k0, hf);
#pragma unroll
      for (int sn = 0; sn < 4; ++sn)
        acc[sn] = wmma_bf16(a, frag_b(cur, 32, wn + sn * 16 + ln, 0, hf), acc[sn]);
      __syncthreads();
    }
#pragma unroll
    for (int sn = 0; sn < 4; ++sn)
#pragma unroll
      for (int r = 0; r < 8; ++r) {
        int row = wm + r + 8 * hf;
        int col = nc * 128 + wn + sn * 16 + ln;
        out[((size_t)n * d1_ + m0 + row) * d2_ + col] =
            acc[sn][r] + bias[(size_t)(m0 + row) * d2_ + col];
      }
  }
}

extern "C" void kernel_launch(void* const* d_in, const int* in_sizes, int n_in,
                              void* d_out, int out_size, void* d_ws, size_t ws_size,
                              hipStream_t stream) {
  (void)in_sizes; (void)n_in; (void)out_size; (void)ws_size;
  const float* x     = (const float*)d_in[0];
  const float* W1    = (const float*)d_in[1];
  const float* W2    = (const float*)d_in[2];
  const float* W     = (const float*)d_in[3];
  const float* alpha = (const float*)d_in[4];
  const float* bias  = (const float*)d_in[5];
  float* out = (float*)d_out;

  char* ws = (char*)d_ws;
  __bf16* WmT  = (__bf16*)ws;                         // 512 KB
  __bf16* W2T  = (__bf16*)(ws + 512 * 1024);          // 256 KB
  __bf16* W1T  = (__bf16*)(ws + 768 * 1024);          // 1 MB
  __bf16* hbuf = (__bf16*)(ws + 1792 * 1024);         // 64 MiB

  tabl_prep_kernel<<<dim3(256), dim3(256), 0, stream>>>(W, W2, W1, WmT, W2T, W1T);
  tabl_gemm1_kernel<<<dim3(4, 4, 128), dim3(256), 0, stream>>>(x, W1T, hbuf);

  (void)hipFuncSetAttribute((const void*)tabl_fused_kernel,
                            hipFuncAttributeMaxDynamicSharedMemorySize, FK_SMEM);
  tabl_fused_kernel<<<dim3(8, 128), dim3(256), FK_SMEM, stream>>>(hbuf, WmT, W2T, alpha, bias, out);
}
